// MocoLoss_18373870092471
// MI455X (gfx1250) — matrix operand order repaired and checked by reading, not compiled, not run on hardware
//
#include <hip/hip_runtime.h>
#include <hip/hip_bf16.h>

// ---------------------------------------------------------------------------
// MoCo contrastive loss for MI455X (gfx1250, wave32, WMMA + async-to-LDS)
// ---------------------------------------------------------------------------

typedef __attribute__((ext_vector_type(16))) __bf16 v16bfx;
typedef __attribute__((ext_vector_type(2)))  __bf16 v2bfx;
typedef __attribute__((ext_vector_type(8)))  float  v8fx;

#define D_DIM   16384
#define B_ROWS  512
#define M_ROWS  1056
#define TEMP_INV (1.0f / 0.07f)
#define EPS_N    1e-8f

// workspace layout (float element offsets)
#define WS_S     0                       // S[512][1056] scaled logits
#define WS_INVQ  (B_ROWS * M_ROWS)       // 512
#define WS_INVM  (WS_INVQ + B_ROWS)      // 1056
#define WS_LPOS  (WS_INVM + M_ROWS)      // 512
#define WS_RLOSS (WS_LPOS + B_ROWS)      // 512
#define WS_TOTAL_F (WS_RLOSS + B_ROWS)   // 543,264 floats (~2.17 MB)
// optional bf16 mirrors (fast path), byte offsets:
#define WS_QB_BYTE ((size_t)WS_TOTAL_F * 4)
#define WS_MB_BYTE (WS_QB_BYTE + (size_t)B_ROWS * D_DIM * 2)
#define WS_NEED_FAST (WS_MB_BYTE + (size_t)M_ROWS * D_DIM * 2)  // ~51.1 MB

// ---- helpers --------------------------------------------------------------

#if __has_builtin(__builtin_amdgcn_sched_barrier)
#define SCHED_FENCE() __builtin_amdgcn_sched_barrier(0)
#else
#define SCHED_FENCE()
#endif

#if __has_builtin(__builtin_amdgcn_cvt_pk_bf16_f32)
__device__ __forceinline__ unsigned pack_bf16x2(float x, float y) {
    v2bfx r = __builtin_amdgcn_cvt_pk_bf16_f32(x, y);
    return __builtin_bit_cast(unsigned, r);
}
#else
// CDNA5 ISA 15.10: V_FMA_MIXLO_BF16 / V_FMA_MIXHI_BF16 (VOP3P op 62/63):
// fma(x*1.0+0) cvt'd RNE to bf16 in the low/high half of the dest dword.
__device__ __forceinline__ unsigned pack_bf16x2(float x, float y) {
    unsigned r;
    asm("v_fma_mixlo_bf16 %0, %1, 1.0, 0\n\t"
        "v_fma_mixhi_bf16 %0, %2, 1.0, 0"
        : "=&v"(r)
        : "v"(x), "v"(y));
    return r;
}
#endif

union FragU { uint4 u[2]; v16bfx v; };

__device__ __forceinline__ uint4 pack8(const float4& p, const float4& q) {
    return make_uint4(pack_bf16x2(p.x, p.y), pack_bf16x2(p.z, p.w),
                      pack_bf16x2(q.x, q.y), pack_bf16x2(q.z, q.w));
}

// LDS byte offset of a shared-memory pointer: flat aperture keeps the LDS
// offset in addr[31:0] (ISA 10.2), matching the LDS byte address the async copy takes from its VGPR.
__device__ __forceinline__ unsigned lds_off(const void* p) {
    return (unsigned)(uintptr_t)p;
}

// Async 16B global -> LDS copy (ISA 15.18.3 GLOBAL_LOAD_ASYNC_TO_LDS_B128,
// tracked by ASYNCcnt).
__device__ __forceinline__ void async_copy16(unsigned lds, const void* gaddr) {
    asm volatile("global_load_async_to_lds_b128 %0, %1, off"
                 :: "v"(lds), "v"(gaddr) : "memory");
}

__device__ __forceinline__ void wait_async0() {
    asm volatile("s_wait_asynccnt 0x0" ::: "memory");
}

// ---- kernel 1: norms + l_pos (+ optional bf16 mirror write) ---------------

__global__ __launch_bounds__(256)
void moco_norms(const float* __restrict__ q, const float* __restrict__ k,
                const float* __restrict__ queue, float* __restrict__ ws,
                unsigned short* __restrict__ qb, unsigned short* __restrict__ mb,
                int conv) {
    __shared__ float red0[256];
    __shared__ float red1[256];
    __shared__ float red2[256];
    const int b = blockIdx.x;
    const int t = threadIdx.x;

    if (b < B_ROWS) {
        const float4* qr = (const float4*)(q + (size_t)b * D_DIM);
        const float4* kr = (const float4*)(k + (size_t)b * D_DIM);
        unsigned short* qo = qb + (size_t)b * D_DIM;
        float sqq = 0.f, skk = 0.f, sqk = 0.f;
        for (int i = t; i < D_DIM / 8; i += 256) {
            float4 q0 = qr[2*i], q1 = qr[2*i+1];
            float4 k0 = kr[2*i], k1 = kr[2*i+1];
            sqq += q0.x*q0.x + q0.y*q0.y + q0.z*q0.z + q0.w*q0.w
                 + q1.x*q1.x + q1.y*q1.y + q1.z*q1.z + q1.w*q1.w;
            skk += k0.x*k0.x + k0.y*k0.y + k0.z*k0.z + k0.w*k0.w
                 + k1.x*k1.x + k1.y*k1.y + k1.z*k1.z + k1.w*k1.w;
            sqk += q0.x*k0.x + q0.y*k0.y + q0.z*k0.z + q0.w*k0.w
                 + q1.x*k1.x + q1.y*k1.y + q1.z*k1.z + q1.w*k1.w;
            if (conv) *(uint4*)(qo + 8 * i) = pack8(q0, q1);
        }
        red0[t] = sqq; red1[t] = skk; red2[t] = sqk;
        __syncthreads();
        for (int s = 128; s > 0; s >>= 1) {
            if (t < s) { red0[t] += red0[t+s]; red1[t] += red1[t+s]; red2[t] += red2[t+s]; }
            __syncthreads();
        }
        if (t == 0) {
            float nq = fmaxf(sqrtf(red0[0]), EPS_N);
            float nk = fmaxf(sqrtf(red1[0]), EPS_N);
            ws[WS_INVQ + b] = 1.0f / nq;
            ws[WS_LPOS + b] = (red2[0] / (nq * nk)) * TEMP_INV;
        }
    } else {
        const int m = b - B_ROWS;
        const float4* mr = (const float4*)(queue + (size_t)m * D_DIM);
        unsigned short* mo = mb + (size_t)m * D_DIM;
        float s = 0.f;
        for (int i = t; i < D_DIM / 8; i += 256) {
            float4 v0 = mr[2*i], v1 = mr[2*i+1];
            s += v0.x*v0.x + v0.y*v0.y + v0.z*v0.z + v0.w*v0.w
               + v1.x*v1.x + v1.y*v1.y + v1.z*v1.z + v1.w*v1.w;
            if (conv) *(uint4*)(mo + 8 * i) = pack8(v0, v1);
        }
        red0[t] = s;
        __syncthreads();
        for (int sft = 128; sft > 0; sft >>= 1) {
            if (t < sft) red0[t] += red0[t+sft];
            __syncthreads();
        }
        if (t == 0) ws[WS_INVM + m] = 1.0f / fmaxf(sqrtf(red0[0]), EPS_N);
    }
}

// ---- GEMM common geometry -------------------------------------------------
// Block tile 64(M)x128(N), K-step 64, double-buffered bf16 tiles in LDS.
// 8 wave32s; each wave owns 32x32 of C: 2 A-frags x 2 B-frags ->
// 4x v_wmma_f32_16x16x32_bf16 per 32-K step, 8 per iteration.

#define BK     64
#define LDSTR  72   // padded row stride in bf16 elems (144B: 16B-aligned rows,
                    // start banks 4*(9*i mod 16): conflict-free)
#define ABUF_BYTES (64 * LDSTR * 2)
#define BBUF_BYTES (128 * LDSTR * 2)

// shared epilogue: acc element j -> M = 8*hi + j, N = lo (16x16 f32 D layout)
__device__ __forceinline__ void gemm_epilogue(
    v8fx acc[2][2], float* __restrict__ ws,
    int bm0, int bn0, int wm, int wn, int lo, int hi) {
    float* S = ws + WS_S;
#pragma unroll
    for (int sn = 0; sn < 2; ++sn) {
        const int gn = bn0 + wn * 32 + sn * 16 + lo;
        if (gn < M_ROWS) {
            const float invn = ws[WS_INVM + gn] * TEMP_INV;
#pragma unroll
            for (int sm = 0; sm < 2; ++sm) {
                const int gmb = bm0 + wm * 32 + sm * 16 + 8 * hi;
#pragma unroll
                for (int j = 0; j < 8; ++j) {
                    const int gm = gmb + j;
                    S[(size_t)gm * M_ROWS + gn] = acc[sm][sn][j] * ws[WS_INVQ + gm] * invn;
                }
            }
        }
    }
}

// fragment math from one LDS buffer (8 WMMAs per call).
// A sched fence ends each 32-K step so next-step ds_loads are not hoisted
// into this step's WMMA group: keeps peak VGPR pressure low enough that the
// accumulators stay pinned (no live-range splits -> no WMMA->VALU v_nops).
__device__ __forceinline__ void gemm_math(
    const unsigned short* asr, const unsigned short* bsr,
    v8fx acc[2][2], int wm, int wn, int lo, int hi) {
#pragma unroll
    for (int kk = 0; kk < BK; kk += 32) {
        FragU fa0, fa1, fb0, fb1;
        const unsigned short* pa = asr + (wm * 32 + lo) * LDSTR + kk + hi * 8;
        fa0.u[0] = *(const uint4*)(pa);
        fa0.u[1] = *(const uint4*)(pa + 16);
        fa1.u[0] = *(const uint4*)(pa + 16 * LDSTR);
        fa1.u[1] = *(const uint4*)(pa + 16 * LDSTR + 16);

        const unsigned short* pb = bsr + (wn * 32 + lo) * LDSTR + kk + hi * 8;
        fb0.u[0] = *(const uint4*)(pb);
        fb0.u[1] = *(const uint4*)(pb + 16);
        fb1.u[0] = *(const uint4*)(pb + 16 * LDSTR);
        fb1.u[1] = *(const uint4*)(pb + 16 * LDSTR + 16);

        acc[0][0] = __builtin_amdgcn_wmma_f32_16x16x32_bf16(
            false, fa0.v, false, fb0.v, (short)0, acc[0][0], false, false);
        acc[0][1] = __builtin_amdgcn_wmma_f32_16x16x32_bf16(
            false, fa0.v, false, fb1.v, (short)0, acc[0][1], false, false);
        acc[1][0] = __builtin_amdgcn_wmma_f32_16x16x32_bf16(
            false, fa1.v, false, fb0.v, (short)0, acc[1][0], false, false);
        acc[1][1] = __builtin_amdgcn_wmma_f32_16x16x32_bf16(
            false, fa1.v, false, fb1.v, (short)0, acc[1][1], false, false);

        SCHED_FENCE();
    }
}

// ---- kernel 2a (fast): bf16 global + async global->LDS --------------------

__global__ __launch_bounds__(256, 1)
void moco_gemm_async(const unsigned short* __restrict__ qb,
                     const unsigned short* __restrict__ mb,
                     float* __restrict__ ws) {
    __shared__ __align__(16) unsigned short As[2 * 64 * LDSTR];
    __shared__ __align__(16) unsigned short Bs[2 * 128 * LDSTR];

    const int t    = threadIdx.x;
    const int lane = t & 31;
    const int wave = t >> 5;
    const int lo   = lane & 15;
    const int hi   = lane >> 4;
    const int wn   = wave & 3;
    const int wm   = wave >> 2;

    const int bm0 = blockIdx.x * 64;
    const int bn0 = blockIdx.y * 128;

    // per-thread async chunks (16B each, 8 bf16):
    // A tile 64x64: 512 chunks -> 2/thread ; B tile 128x64: 1024 -> 4/thread.
    unsigned aL[2]; const unsigned short* aG[2];
#pragma unroll
    for (int i = 0; i < 2; ++i) {
        const int c  = t + 256 * i;
        const int r  = c >> 3;            // 0..63
        const int co = (c & 7) * 8;       // bf16 elems within K-step
        aL[i] = lds_off(&As[r * LDSTR + co]);
        aG[i] = qb + (size_t)(bm0 + r) * D_DIM + co;
    }
    unsigned bL[4]; const unsigned short* bG[4];
#pragma unroll
    for (int i = 0; i < 4; ++i) {
        const int c  = t + 256 * i;
        const int r  = c >> 3;            // 0..127
        const int co = (c & 7) * 8;
        int gr = bn0 + r; if (gr > M_ROWS - 1) gr = M_ROWS - 1;  // clamp; store guarded
        bL[i] = lds_off(&Bs[r * LDSTR + co]);
        bG[i] = mb + (size_t)gr * D_DIM + co;
    }

    v8fx acc[2][2] = {};

    // prologue: tile 0 -> buffer 0
#pragma unroll
    for (int i = 0; i < 2; ++i) async_copy16(aL[i], aG[i]);
#pragma unroll
    for (int i = 0; i < 4; ++i) async_copy16(bL[i], bG[i]);
    wait_async0();
    __syncthreads();

    const int NT = D_DIM / BK;   // 256 K-steps
    for (int kt = 0; kt < NT; ++kt) {
        // issue async copies for the next tile into the other buffer
        if (kt + 1 < NT) {
            const unsigned selA = (unsigned)((kt + 1) & 1) * ABUF_BYTES;
            const unsigned selB = (unsigned)((kt + 1) & 1) * BBUF_BYTES;
            const size_t koff = (size_t)(kt + 1) * BK;
#pragma unroll
            for (int i = 0; i < 2; ++i) async_copy16(aL[i] + selA, aG[i] + koff);
#pragma unroll
            for (int i = 0; i < 4; ++i) async_copy16(bL[i] + selB, bG[i] + koff);
        }

        // math from current buffer
        gemm_math(As + (kt & 1) * 64 * LDSTR, Bs + (kt & 1) * 128 * LDSTR,
                  acc, wm, wn, lo, hi);

        wait_async0();     // own copies for next tile done (hidden under math)
        __syncthreads();   // everyone's copies visible; readers of old buf done
    }

    gemm_epilogue(acc, ws, bm0, bn0, wm, wn, lo, hi);
}

// ---- kernel 2b (fallback): f32 global, fused convert ----------------------

__global__ __launch_bounds__(256, 1)
void moco_gemm(const float* __restrict__ q, const float* __restrict__ queue,
               float* __restrict__ ws) {
    __shared__ __align__(16) unsigned short As[2 * 64 * LDSTR];
    __shared__ __align__(16) unsigned short Bs[2 * 128 * LDSTR];

    const int t    = threadIdx.x;
    const int lane = t & 31;
    const int wave = t >> 5;
    const int lo   = lane & 15;
    const int hi   = lane >> 4;
    const int wn   = wave & 3;
    const int wm   = wave >> 2;

    const int bm0 = blockIdx.x * 64;
    const int bn0 = blockIdx.y * 128;

    const int lrA = t >> 2, lcA = (t & 3) * 16;
    const int lrB = t >> 1, lcB = (t & 1) * 32;
    const float* aptr = q + (size_t)(bm0 + lrA) * D_DIM + lcA;
    int brow = bn0 + lrB; if (brow > M_ROWS - 1) brow = M_ROWS - 1;
    const float* bptr = queue + (size_t)brow * D_DIM + lcB;

    v8fx acc[2][2] = {};

    float4 a[4], b[8];
#pragma unroll
    for (int i = 0; i < 4; ++i) a[i] = *(const float4*)(aptr + 4 * i);
#pragma unroll
    for (int i = 0; i < 8; ++i) b[i] = *(const float4*)(bptr + 4 * i);

    const int NT = D_DIM / BK;
    for (int kt = 0; kt < NT; ++kt) {
        const int bufA = (kt & 1) * 64 * LDSTR;
        const int bufB = (kt & 1) * 128 * LDSTR;
        unsigned short* as = As + bufA + lrA * LDSTR + lcA;
        *(uint4*)(as)     = pack8(a[0], a[1]);
        *(uint4*)(as + 8) = pack8(a[2], a[3]);
        unsigned short* bs = Bs + bufB + lrB * LDSTR + lcB;
        *(uint4*)(bs)      = pack8(b[0], b[1]);
        *(uint4*)(bs + 8)  = pack8(b[2], b[3]);
        *(uint4*)(bs + 16) = pack8(b[4], b[5]);
        *(uint4*)(bs + 24) = pack8(b[6], b[7]);
        __syncthreads();

        if (kt + 1 < NT) {
            const float* ap = aptr + (size_t)(kt + 1) * BK;
            const float* bp = bptr + (size_t)(kt + 1) * BK;
#pragma unroll
            for (int i = 0; i < 4; ++i) a[i] = *(const float4*)(ap + 4 * i);
#pragma unroll
            for (int i = 0; i < 8; ++i) b[i] = *(const float4*)(bp + 4 * i);
        }

        gemm_math(As + bufA, Bs + bufB, acc, wm, wn, lo, hi);
        __syncthreads();
    }

    gemm_epilogue(acc, ws, bm0, bn0, wm, wn, lo, hi);
}

// ---- kernel 3: per-row logsumexp loss -------------------------------------

__global__ __launch_bounds__(256)
void moco_rowloss(float* __restrict__ ws) {
    __shared__ float red[256];
    const int i = blockIdx.x;
    const int t = threadIdx.x;
    const float4* row = (const float4*)(ws + WS_S + (size_t)i * M_ROWS);
    const float lp = ws[WS_LPOS + i];

    float mx = lp;
    for (int j = t; j < M_ROWS / 4; j += 256) {
        float4 v = row[j];
        mx = fmaxf(mx, fmaxf(fmaxf(v.x, v.y), fmaxf(v.z, v.w)));
    }
    red[t] = mx;
    __syncthreads();
    for (int s = 128; s > 0; s >>= 1) {
        if (t < s) red[t] = fmaxf(red[t], red[t + s]);
        __syncthreads();
    }
    mx = red[0];
    __syncthreads();

    float sum = 0.f;
    for (int j = t; j < M_ROWS / 4; j += 256) {
        float4 v = row[j];
        sum += expf(v.x - mx) + expf(v.y - mx) + expf(v.z - mx) + expf(v.w - mx);
    }
    red[t] = sum;
    __syncthreads();
    for (int s = 128; s > 0; s >>= 1) {
        if (t < s) red[t] += red[t + s];
        __syncthreads();
    }
    if (t == 0) {
        float tot = red[0] + expf(lp - mx);
        ws[WS_RLOSS + i] = (logf(tot) + mx) - lp;   // = logsumexp - out[i,0]
    }
}

// ---- kernel 4: mean over rows ---------------------------------------------

__global__ __launch_bounds__(512)
void moco_reduce(const float* __restrict__ ws, float* __restrict__ out) {
    __shared__ float red[512];
    const int t = threadIdx.x;
    red[t] = ws[WS_RLOSS + t];
    __syncthreads();
    for (int s = 256; s > 0; s >>= 1) {
        if (t < s) red[t] += red[t + s];
        __syncthreads();
    }
    if (t == 0) out[0] = red[0] / (float)B_ROWS;
}

// ---------------------------------------------------------------------------

extern "C" void kernel_launch(void* const* d_in, const int* in_sizes, int n_in,
                              void* d_out, int out_size, void* d_ws, size_t ws_size,
                              hipStream_t stream) {
    const float* feat_q = (const float*)d_in[0];
    const float* feat_k = (const float*)d_in[1];
    const float* queue  = (const float*)d_in[2];
    // d_in[3] = idx (int64): unused by the loss, matches reference
    float* ws  = (float*)d_ws;
    float* out = (float*)d_out;

    // deterministic path choice: bf16 mirrors + async-to-LDS GEMM if the
    // workspace can hold them, else fused-convert GEMM from f32 inputs.
    const int fast = (ws_size >= WS_NEED_FAST) ? 1 : 0;
    unsigned short* qb = (unsigned short*)((char*)d_ws + WS_QB_BYTE);
    unsigned short* mb = (unsigned short*)((char*)d_ws + WS_MB_BYTE);

    moco_norms<<<dim3(B_ROWS + M_ROWS), 256, 0, stream>>>(
        feat_q, feat_k, queue, ws, qb, mb, fast);
    if (fast)
        moco_gemm_async<<<dim3(B_ROWS / 64, (M_ROWS + 127) / 128), 256, 0, stream>>>(qb, mb, ws);
    else
        moco_gemm<<<dim3(B_ROWS / 64, (M_ROWS + 127) / 128), 256, 0, stream>>>(feat_q, queue, ws);
    moco_rowloss<<<dim3(B_ROWS), 256, 0, stream>>>(ws);
    moco_reduce <<<1, 512, 0, stream>>>(ws, out);
}
